// SpatialMPNN_46342697124300
// MI455X (gfx1250) — matrix operand order repaired and checked by reading, not compiled, not run on hardware
//
#include <hip/hip_runtime.h>

typedef __attribute__((ext_vector_type(16))) _Float16 v16h;
typedef __attribute__((ext_vector_type(8)))  float    v8f;
typedef __attribute__((ext_vector_type(4)))  float    v4f;

#define N_NODES 50000
#define N_EDGES 640000
#define HIDDEN  128
#define NLAYERS 4
#define NUM_RBF 50
#define NGRAPHS 64
#define LN_EPS  1e-5f
#define CUTOFF  4.0f

union H2U { _Float16 h[2]; unsigned int u; };
union V16 { uint4 q[2]; v16h v; };

__device__ __forceinline__ v8f wmma_f16(v16h a, v16h b, v8f c) {
  // D = A(16x32 f16) * B(32x16 f16) + C(16x16 f32)
  return __builtin_amdgcn_wmma_f32_16x16x32_f16(false, a, false, b, (short)0, c, false, false);
}

// ---------------------------------------------------------------------------
// Pre-swizzle fw1 (pad K 50->64), fw2, lw into WMMA B-operand lane layout.
// Per 16x16x32 B tile: lane L -> N = nt*16 + (L&15), kgroup = L>>4,
// dword i holds halves (K = kt*32 + kg*8 + 2i for i<4, kt*32 + 16 + kg*8 + 2(i-4) for i>=4, +1). Stored lane-major,
// 32 lanes x 8 dwords = 1 KB per tile.
// wsw dword layout: fw1s [0,16384)  fw2s [16384,49152)  lws [49152,81920)
// Per-layer regions are contiguous: fw1 layer l = wsw + l*4096 (4096 dwords),
// fw2 layer l = wsw + 16384 + l*8192, lw layer l = wsw + 49152 + l*8192.
// ---------------------------------------------------------------------------
__global__ void wprep(const float* __restrict__ fw1, const float* __restrict__ fw2,
                      const float* __restrict__ lw, unsigned int* __restrict__ wsw) {
  int tile = blockIdx.x;
  int lane = threadIdx.x;
  const float* src; int Ksrc; unsigned int* dst; int kt, nt;
  if (tile < 64) {                       // fw1: 4 layers x 2 ktiles x 8 ntiles
    int t = tile; int layer = t >> 4; int r = t & 15; kt = r >> 3; nt = r & 7;
    src = fw1 + layer * NUM_RBF * HIDDEN; Ksrc = NUM_RBF;
    dst = wsw + t * 256;
  } else if (tile < 192) {               // fw2: 4 x 4 x 8
    int t = tile - 64; int layer = t >> 5; int r = t & 31; kt = r >> 3; nt = r & 7;
    src = fw2 + layer * HIDDEN * HIDDEN; Ksrc = HIDDEN;
    dst = wsw + 16384 + t * 256;
  } else {                               // lw: 4 x 4 x 8
    int t = tile - 192; int layer = t >> 5; int r = t & 31; kt = r >> 3; nt = r & 7;
    src = lw + layer * HIDDEN * HIDDEN; Ksrc = HIDDEN;
    dst = wsw + 49152 + t * 256;
  }
  int n  = nt * 16 + (lane & 15);
  int kg = lane >> 4;
#pragma unroll
  for (int i = 0; i < 8; ++i) {
    int k0 = kt * 32 + ((i < 4) ? (kg * 8 + 2 * i) : (16 + kg * 8 + 2 * (i - 4)));
    float a0 = (k0     < Ksrc) ? src[k0 * HIDDEN + n]       : 0.f;
    float a1 = (k0 + 1 < Ksrc) ? src[(k0 + 1) * HIDDEN + n] : 0.f;
    H2U p; p.h[0] = (_Float16)a0; p.h[1] = (_Float16)a1;
    dst[lane * 8 + i] = p.u;
  }
}

// h = emb[x], out = 0
__global__ void embed_init(const int* __restrict__ x, const float* __restrict__ emb,
                           float* __restrict__ h, float* __restrict__ outacc) {
  int idx = blockIdx.x * blockDim.x + threadIdx.x;
  if (idx >= N_NODES * HIDDEN) return;
  int n = idx >> 7, k = idx & 127;
  h[idx] = emb[x[n] * HIDDEN + k];
  outacc[idx] = 0.f;
}

// xl = h @ lw[layer] + lb[layer]   (wave per 16-node tile, WMMA, B from LDS)
// dynamic LDS: 32 KB = lw layer tiles (8192 dwords)
__global__ void __launch_bounds__(256) node_gemm(const float* __restrict__ hbuf,
                          const unsigned int* __restrict__ wsw,
                          const float* __restrict__ lb, float* __restrict__ xl, int layer) {
  extern __shared__ __align__(16) char dynlds[];
  unsigned int* wl = (unsigned int*)dynlds;                 // 8192 dwords

  // cooperative stage: lw layer region -> LDS (2048 uint4 / 256 threads)
  {
    const uint4* g = (const uint4*)(wsw + 49152 + layer * 8192);
    uint4* l = (uint4*)wl;
    int tid = threadIdx.x;
#pragma unroll
    for (int i = 0; i < 8; ++i) l[tid + i * 256] = g[tid + i * 256];
  }
  __syncthreads();

  int wave = (blockIdx.x * blockDim.x + threadIdx.x) >> 5;
  int lane = threadIdx.x & 31;
  if (wave >= N_NODES / 16) return;
  int n0 = wave * 16;
  int M = lane & 15, s = lane >> 4;

  // A tiles: row M of h, f32 -> f16.  A layout: elems 0..7 = K kb+s*8+j, 8..15 = K kb+16+s*8+j
  v16h a[4];
  const float* row = hbuf + (size_t)(n0 + M) * HIDDEN;
#pragma unroll
  for (int kt = 0; kt < 4; ++kt) {
    int klo = kt * 32 + s * 8;
    v4f f0 = *(const v4f*)(row + klo);
    v4f f1 = *(const v4f*)(row + klo + 4);
    v4f f2 = *(const v4f*)(row + klo + 16);
    v4f f3 = *(const v4f*)(row + klo + 20);
    v16h av;
#pragma unroll
    for (int j = 0; j < 4; ++j) {
      av[j] = (_Float16)f0[j]; av[4 + j] = (_Float16)f1[j];
      av[8 + j] = (_Float16)f2[j]; av[12 + j] = (_Float16)f3[j];
    }
    a[kt] = av;
  }

  const float* bias = lb + layer * HIDDEN;
#pragma unroll
  for (int nt = 0; nt < 8; ++nt) {
    // preload all 4 B tiles, then run the WMMA chain
    V16 bv[4];
#pragma unroll
    for (int kt = 0; kt < 4; ++kt) {
      const unsigned int* bt = wl + ((kt * 8 + nt) * 256) + lane * 8;
      bv[kt].q[0] = *(const uint4*)bt; bv[kt].q[1] = *(const uint4*)(bt + 4);
    }
    float b = bias[nt * 16 + M];
    v8f c;
#pragma unroll
    for (int r = 0; r < 8; ++r) c[r] = b;
#pragma unroll
    for (int kt = 0; kt < 4; ++kt) c = wmma_f16(a[kt], bv[kt].v, c);
#pragma unroll
    for (int r = 0; r < 8; ++r)       // C layout: vgpr r -> row r+8s, lane -> col
      xl[(size_t)(n0 + r + 8 * s) * HIDDEN + nt * 16 + M] = c[r];
  }
}

// Per 16-edge tile: W = relu(rbf@fw1+fb1)@fw2+fb2 ; msg = xl[col]*W ; atomic out[row] += msg
// dynamic LDS (80 KB): [0,32K) relu slots (8 waves x 4KB)
//                      [32K,48K) fw1 layer tiles   [48K,80K) fw2 layer tiles
__global__ void __launch_bounds__(256) edge_msg(const float* __restrict__ eattr,
      const int* __restrict__ eidx, const float* __restrict__ xl,
      float* __restrict__ outacc, const unsigned int* __restrict__ wsw,
      const float* __restrict__ fb1, const float* __restrict__ fb2, int layer) {
  extern __shared__ __align__(16) char dynlds[];
  _Float16*     sm = (_Float16*)dynlds;                     // 16K halfs
  unsigned int* l1 = (unsigned int*)(dynlds + 32768);       // 4096 dwords (fw1)
  unsigned int* l2 = (unsigned int*)(dynlds + 49152);       // 8192 dwords (fw2)

  // cooperative stage of this layer's weight tiles into LDS
  {
    int tid = threadIdx.x;
    const uint4* g1 = (const uint4*)(wsw + layer * 4096);
    const uint4* g2 = (const uint4*)(wsw + 16384 + layer * 8192);
    uint4* d1 = (uint4*)l1;
    uint4* d2 = (uint4*)l2;
#pragma unroll
    for (int i = 0; i < 4; ++i) d1[tid + i * 256] = g1[tid + i * 256];
#pragma unroll
    for (int i = 0; i < 8; ++i) d2[tid + i * 256] = g2[tid + i * 256];
  }
  __syncthreads();

  int warp = threadIdx.x >> 5, lane = threadIdx.x & 31;
  int tile = blockIdx.x * 8 + warp;
  int e0 = tile * 16;
  int M = lane & 15, s = lane >> 4;
  _Float16* slot = sm + warp * (16 * HIDDEN);

  const float delta = CUTOFF / (float)(NUM_RBF - 1);
  const float coeff = -0.5f / (delta * delta);
  float dist = eattr[e0 + M];

  // A1: gaussian RBF generated directly in WMMA A layout (K padded to 64)
  v16h a1[2];
#pragma unroll
  for (int kt = 0; kt < 2; ++kt) {
    v16h av;
#pragma unroll
    for (int j = 0; j < 16; ++j) {
      int k = kt * 32 + ((j < 8) ? (s * 8 + j) : (16 + s * 8 + (j - 8)));
      float v = 0.f;
      if (k < NUM_RBF) { float d = dist - (float)k * delta; v = __expf(coeff * d * d); }
      av[j] = (_Float16)v;
    }
    a1[kt] = av;
  }

  // GEMM1 + ReLU -> LDS (row-major [16 edges][128])
  const float* b1 = fb1 + layer * HIDDEN;
#pragma unroll
  for (int nt = 0; nt < 8; ++nt) {
    V16 bv[2];
#pragma unroll
    for (int kt = 0; kt < 2; ++kt) {
      const unsigned int* bt = l1 + ((kt * 8 + nt) * 256) + lane * 8;
      bv[kt].q[0] = *(const uint4*)bt; bv[kt].q[1] = *(const uint4*)(bt + 4);
    }
    float b = b1[nt * 16 + M];
    v8f c;
#pragma unroll
    for (int r = 0; r < 8; ++r) c[r] = b;
#pragma unroll
    for (int kt = 0; kt < 2; ++kt) c = wmma_f16(a1[kt], bv[kt].v, c);
#pragma unroll
    for (int r = 0; r < 8; ++r)
      slot[(r + 8 * s) * HIDDEN + nt * 16 + M] = (_Float16)fmaxf(c[r], 0.f);
  }
  asm volatile("s_wait_dscnt 0" ::: "memory");   // intra-wave LDS RAW

  // A2 tiles from LDS (transpose C layout -> A layout)
  v16h a2[4];
#pragma unroll
  for (int kt = 0; kt < 4; ++kt) {
    int klo = kt * 32 + s * 8;
    const _Float16* p = slot + M * HIDDEN + klo;
    V16 av; av.q[0] = *(const uint4*)p; av.q[1] = *(const uint4*)(p + 16);
    a2[kt] = av.v;
  }

  // edge indices for this lane's 8 C-rows (e = e0 + r + 8s)
  int ridx[8], cidx[8];
#pragma unroll
  for (int r = 0; r < 8; ++r) {
    int e = e0 + r + 8 * s;
    ridx[r] = eidx[e] * HIDDEN;            // scatter target (row)
    cidx[r] = eidx[N_EDGES + e] * HIDDEN;  // gather source (col)
  }

  const float* b2 = fb2 + layer * HIDDEN;
#pragma unroll
  for (int nt = 0; nt < 8; ++nt) {
    V16 bv[4];
#pragma unroll
    for (int kt = 0; kt < 4; ++kt) {
      const unsigned int* bt = l2 + ((kt * 8 + nt) * 256) + lane * 8;
      bv[kt].q[0] = *(const uint4*)bt; bv[kt].q[1] = *(const uint4*)(bt + 4);
    }
    float b = b2[nt * 16 + M];
    v8f c;
#pragma unroll
    for (int r = 0; r < 8; ++r) c[r] = b;
#pragma unroll
    for (int kt = 0; kt < 4; ++kt) c = wmma_f16(a2[kt], bv[kt].v, c);

    int hcol = nt * 16 + M;
#pragma unroll
    for (int r = 0; r < 8; ++r) {
      float msg = xl[cidx[r] + hcol] * c[r];
      atomicAdd(outacc + ridx[r] + hcol, msg);
    }
  }
}

// h = LayerNorm(h + out) ; out = 0    (wave per node)
__global__ void ln_kernel(float* __restrict__ hbuf, float* __restrict__ outacc,
                          const float* __restrict__ ln_g, const float* __restrict__ ln_b,
                          int layer) {
  int wave = (blockIdx.x * blockDim.x + threadIdx.x) >> 5;
  int lane = threadIdx.x & 31;
  if (wave >= N_NODES) return;
  float* hp = hbuf + (size_t)wave * HIDDEN;
  float* op = outacc + (size_t)wave * HIDDEN;
  float v[4], sum = 0.f;
#pragma unroll
  for (int j = 0; j < 4; ++j) { int k = lane + 32 * j; v[j] = hp[k] + op[k]; sum += v[j]; }
#pragma unroll
  for (int d = 16; d >= 1; d >>= 1) sum += __shfl_xor(sum, d, 32);
  float mu = sum * (1.f / HIDDEN);
  float sq = 0.f;
#pragma unroll
  for (int j = 0; j < 4; ++j) { float t = v[j] - mu; sq += t * t; }
#pragma unroll
  for (int d = 16; d >= 1; d >>= 1) sq += __shfl_xor(sq, d, 32);
  float rs = rsqrtf(sq * (1.f / HIDDEN) + LN_EPS);
  const float* g = ln_g + layer * HIDDEN;
  const float* b = ln_b + layer * HIDDEN;
#pragma unroll
  for (int j = 0; j < 4; ++j) {
    int k = lane + 32 * j;
    hp[k] = (v[j] - mu) * rs * g[k] + b[k];
    op[k] = 0.f;                          // re-arm accumulator for next layer
  }
}

__global__ void pool_zero(float* __restrict__ out, float* __restrict__ cnt) {
  int i = blockIdx.x * blockDim.x + threadIdx.x;
  if (i < NGRAPHS * HIDDEN) out[i] = 0.f;
  if (i < NGRAPHS) cnt[i] = 0.f;
}

__global__ void pool_acc(const float* __restrict__ hbuf, const int* __restrict__ batch,
                         float* __restrict__ out, float* __restrict__ cnt) {
  int wave = (blockIdx.x * blockDim.x + threadIdx.x) >> 5;
  int lane = threadIdx.x & 31;
  if (wave >= N_NODES) return;
  int g = batch[wave];
  const float* hp = hbuf + (size_t)wave * HIDDEN;
  float* op = out + g * HIDDEN;
#pragma unroll
  for (int j = 0; j < 4; ++j) { int k = lane + 32 * j; atomicAdd(op + k, hp[k]); }
  if (lane == 0) atomicAdd(cnt + g, 1.f);
}

__global__ void pool_div(float* __restrict__ out, const float* __restrict__ cnt) {
  int i = blockIdx.x * blockDim.x + threadIdx.x;
  if (i >= NGRAPHS * HIDDEN) return;
  out[i] /= fmaxf(cnt[i >> 7], 1.f);
}

extern "C" void kernel_launch(void* const* d_in, const int* in_sizes, int n_in,
                              void* d_out, int out_size, void* d_ws, size_t ws_size,
                              hipStream_t stream) {
  const int*   x     = (const int*)  d_in[0];
  const int*   ei    = (const int*)  d_in[1];
  const float* ea    = (const float*)d_in[2];
  const int*   batch = (const int*)  d_in[3];
  const float* emb   = (const float*)d_in[4];
  const float* fw1   = (const float*)d_in[5];
  const float* fb1   = (const float*)d_in[6];
  const float* fw2   = (const float*)d_in[7];
  const float* fb2   = (const float*)d_in[8];
  const float* lw    = (const float*)d_in[9];
  const float* lb    = (const float*)d_in[10];
  const float* ln_g  = (const float*)d_in[11];
  const float* ln_b  = (const float*)d_in[12];
  float* out = (float*)d_out;

  char* ws = (char*)d_ws;
  const size_t NH = (size_t)N_NODES * HIDDEN * sizeof(float);   // 25.6 MB
  float*        h    = (float*)(ws);
  float*        xl   = (float*)(ws + NH);
  float*        oacc = (float*)(ws + 2 * NH);
  unsigned int* wsw  = (unsigned int*)(ws + 3 * NH);            // 320 KB swizzled weights
  float*        cnt  = (float*)(ws + 3 * NH + 320 * 1024);

  wprep<<<320, 32, 0, stream>>>(fw1, fw2, lw, wsw);
  embed_init<<<(N_NODES * HIDDEN + 255) / 256, 256, 0, stream>>>(x, emb, h, oacc);

  for (int layer = 0; layer < NLAYERS; ++layer) {
    node_gemm<<<(N_NODES / 16 + 7) / 8, 256, 32 * 1024, stream>>>(h, wsw, lb, xl, layer);
    edge_msg<<<N_EDGES / 16 / 8, 256, 80 * 1024, stream>>>(ea, ei, xl, oacc, wsw, fb1, fb2, layer);
    ln_kernel<<<(N_NODES + 7) / 8, 256, 0, stream>>>(h, oacc, ln_g, ln_b, layer);
  }

  pool_zero<<<(NGRAPHS * HIDDEN + 255) / 256, 256, 0, stream>>>(out, cnt);
  pool_acc<<<(N_NODES + 7) / 8, 256, 0, stream>>>(h, batch, out, cnt);
  pool_div<<<(NGRAPHS * HIDDEN + 255) / 256, 256, 0, stream>>>(out, cnt);
}
